// st_conv_block_68504728371267
// MI455X (gfx1250) — matrix operand order repaired and checked by reading, not compiled, not run on hardware
//
#include <hip/hip_runtime.h>
#include <stdint.h>
#include <string.h>

// ---------------- problem constants ----------------
#define NN   1024            // graph nodes
#define TT   64              // input time steps
#define CC   64              // channels (C0=C1=C2)
#define T1   62              // after conv1 (VALID, KT=3)
#define T2   60              // after conv2
#define NBAT 8               // batch
#define G1C  (NBAT*T1)       // 496 groups for conv1/theta/graph gemms
#define G2C  (NBAT*T2)       // 480 groups for conv2
#define LINF (1<<30)         // "linear group" divisor

// ---------------- CDNA5 async-to-LDS probe ----------------
#if defined(__has_builtin)
#  if __has_builtin(__builtin_amdgcn_global_load_async_to_lds_b128)
#    define HAVE_ASYNC_LDS 1
#  endif
#  if __has_builtin(__builtin_amdgcn_s_wait_asynccnt)
#    define HAVE_WAIT_ASYNC 1
#  endif
#endif

#define AS1 __attribute__((address_space(1)))
#define AS3 __attribute__((address_space(3)))
typedef int v4i __attribute__((ext_vector_type(4)));

__device__ __forceinline__ void wait_asynccnt0() {
#if defined(HAVE_WAIT_ASYNC)
  __builtin_amdgcn_s_wait_asynccnt(0);
#else
  asm volatile("s_wait_asynccnt 0x0" ::: "memory");
#endif
}

// ---------------- bf16 helpers ----------------
typedef __attribute__((ext_vector_type(16))) __bf16 v16bf;
typedef __attribute__((ext_vector_type(8)))  float  v8f;

union FragAB { uint4 q[2]; v16bf v; };
union U32F   { unsigned int u; float f; };

__device__ __forceinline__ float bf2f(unsigned short s) {
  U32F c; c.u = ((unsigned int)s) << 16; return c.f;
}
__device__ __forceinline__ unsigned short f2bf(float f) {
  U32F c; c.f = f;
  unsigned int r = (c.u + 0x7FFFu + ((c.u >> 16) & 1u)) >> 16;
  return (unsigned short)r;
}

// ---------------- generic bf16 WMMA GEMM ----------------
// C[group][M x 1024] = A[group][M x K] * B[group][K x 1024]  (+bias,res,act)
// A: row-major bf16, element (r,k) at A + aBase(g) + r*aRS + k
// B: row-major bf16, element (k,c) at B + bBase(g) + (k>>bShift)*bChunk
//                                      + (k & ((1<<bShift)-1))*bKS + c
// per-group base: (g/GT)*BS + (g%GT)*TS + Cn
// C dest: cBase(g) + (r/cD)*cS1 + (r%cD)*cS2 + c*cSC  (f32 or bf16)
// Block tile: 64 rows x 128 cols, 8 waves, each wave = 16 rows x 64 cols (4 WMMA/K-step)
struct GemmParams {
  const unsigned short* A;
  long aBS, aTS, aCn, aRS; int aGT;
  const unsigned short* B;
  long bBS, bTS, bCn, bChunk, bKS; int bGT, bShift;
  float* Cf; unsigned short* Cb;
  long cBS, cTS, cCn, cS1, cS2, cSC; int cGT, cD;
  const float* bias;
  const unsigned short* R1;
  long r1BS, r1TS, r1Cn, r1RS, r1CS; int r1GT;
  const unsigned short* R2;
  long r2BS, r2TS, r2Cn, r2RS, r2CS; int r2GT;
  int M, K, act, scaleEye;
};

__global__ __launch_bounds__(256)
void gemm_bf16_wmma(GemmParams p) {
  __shared__ unsigned short At[64][40];    // [row][k]  padded stride 80B
  __shared__ unsigned short Bt[128][40];   // [col][k]  (transposed tile)
  const int tid  = threadIdx.x;
  const int lane = tid & 31;
  const int wave = tid >> 5;
  const int wrow = wave & 3;     // 4 row tiles of 16
  const int wcol = wave >> 2;    // 2 col groups of 64
  const int g = blockIdx.z;

  const long aBase = (long)(g / p.aGT) * p.aBS + (long)(g % p.aGT) * p.aTS + p.aCn;
  const long bBase = (long)(g / p.bGT) * p.bBS + (long)(g % p.bGT) * p.bTS + p.bCn;
  const long cBase = (long)(g / p.cGT) * p.cBS + (long)(g % p.cGT) * p.cTS + p.cCn;
  const long r1Base = p.R1 ? (long)(g / p.r1GT) * p.r1BS + (long)(g % p.r1GT) * p.r1TS + p.r1Cn : 0;
  const long r2Base = p.R2 ? (long)(g / p.r2GT) * p.r2BS + (long)(g % p.r2GT) * p.r2TS + p.r2Cn : 0;

  const int rowBase = blockIdx.y * 64;
  const int colBase = blockIdx.x * 128;

  v8f acc[4] = {v8f{}, v8f{}, v8f{}, v8f{}};

  const int laRow = tid >> 2, laSeg = (tid & 3) << 3;   // A loader: 64 rows x 32 k
  const int bMask = (1 << p.bShift) - 1;

  for (int k0 = 0; k0 < p.K; k0 += 32) {
    const unsigned short* aSrc =
        p.A + aBase + (long)(rowBase + laRow) * p.aRS + k0 + laSeg;
#if !defined(HAVE_ASYNC_LDS)
    const uint4 av = *(const uint4*)aSrc;
#endif
    // B: 32 k x 128 cols, two pieces per thread, transpose-on-store
    uint4 bv[2]; int bc8[2], bk[2];
    #pragma unroll
    for (int pc = 0; pc < 2; ++pc) {
      const int idx = tid + pc * 256;           // 0..511
      const int k = idx >> 4;                   // 0..31
      const int c8 = (idx & 15) << 3;           // 0..120
      const int kk = k0 + k;
      const long boff = bBase + (long)(kk >> p.bShift) * p.bChunk
                              + (long)(kk & bMask) * p.bKS + (colBase + c8);
      bv[pc] = *(const uint4*)(p.B + boff);
      bc8[pc] = c8; bk[pc] = k;
    }

    if (k0 + 32 < p.K) {   // prefetch next tiles -> global_prefetch_b8
      __builtin_prefetch(aSrc + 32, 0, 0);
      const int kn = k0 + 32 + (tid >> 4);
      __builtin_prefetch(p.B + bBase + (long)(kn >> p.bShift) * p.bChunk
                              + (long)(kn & bMask) * p.bKS + (colBase + ((tid & 15) << 3)), 0, 0);
    }

    __syncthreads();                       // previous iter frag reads complete
#if defined(HAVE_ASYNC_LDS)
    // CDNA5 async DMA: global -> LDS without touching VGPRs (ASYNCcnt)
    __builtin_amdgcn_global_load_async_to_lds_b128(
        (AS1 v4i*)aSrc, (AS3 v4i*)&At[laRow][laSeg], 0, 0);
#else
    *(uint4*)&At[laRow][laSeg] = av;
#endif
    #pragma unroll
    for (int pc = 0; pc < 2; ++pc) {
      union { uint4 q; unsigned short s[8]; } bu; bu.q = bv[pc];
      #pragma unroll
      for (int j = 0; j < 8; ++j) Bt[bc8[pc] + j][bk[pc]] = bu.s[j];
    }
#if defined(HAVE_ASYNC_LDS)
    wait_asynccnt0();                      // A tile landed in LDS
#endif
    __syncthreads();

    // fragment loads per ISA 7.12.2 layouts (wave32)
    const int r16 = lane & 15, kh = lane >> 4;
    FragAB fa;
    fa.q[0] = *(const uint4*)&At[wrow * 16 + r16][kh * 8];
    fa.q[1] = *(const uint4*)&At[wrow * 16 + r16][16 + kh * 8];
    #pragma unroll
    for (int cf = 0; cf < 4; ++cf) {
      FragAB fb;
      fb.q[0] = *(const uint4*)&Bt[wcol * 64 + cf * 16 + r16][kh * 8];
      fb.q[1] = *(const uint4*)&Bt[wcol * 64 + cf * 16 + r16][16 + kh * 8];
      acc[cf] = __builtin_amdgcn_wmma_f32_16x16x32_bf16(false, fa.v, false, fb.v,
                                                        (short)0, acc[cf], false, false);
    }
  }

  // epilogue
  #pragma unroll
  for (int cf = 0; cf < 4; ++cf) {
    const int gcol = colBase + wcol * 64 + cf * 16 + (lane & 15);
    #pragma unroll
    for (int e = 0; e < 8; ++e) {
      const int grow = rowBase + wrow * 16 + (lane >> 4) * 8 + e;
      float v = acc[cf][e];
      if (p.bias) v += p.bias[grow];
      if (p.R1)   v += bf2f(p.R1[r1Base + (long)grow * p.r1RS + (long)gcol * p.r1CS]);
      if (p.R2)   v += bf2f(p.R2[r2Base + (long)grow * p.r2RS + (long)gcol * p.r2CS]);
      if (p.scaleEye) v = 2.f * v - ((grow == gcol) ? 1.f : 0.f);
      if (p.act)  v = fmaxf(v, 0.f);
      const long dst = cBase + (long)(grow / p.cD) * p.cS1
                             + (long)(grow % p.cD) * p.cS2 + (long)gcol * p.cSC;
      if (p.Cf) p.Cf[dst] = v; else p.Cb[dst] = f2bf(v);
    }
  }
}

// ---------------- small kernels ----------------
__global__ void k_f32_to_bf16(const float* __restrict__ s, unsigned short* __restrict__ d, int n) {
  int i = blockIdx.x * 256 + threadIdx.x; if (i < n) d[i] = f2bf(s[i]);
}
// w1 (128,64,3) -> w1r[co][kt*64+i]
__global__ void k_pack_w1(const float* w, unsigned short* o) {
  int idx = blockIdx.x * 256 + threadIdx.x; if (idx >= 128 * 192) return;
  int co = idx / 192, k = idx % 192, kt = k >> 6, i = k & 63;
  o[idx] = f2bf(w[co * 192 + i * 3 + kt]);
}
// theta (64,64,3) -> thetaE[r=(k*64+o)][i]
__global__ void k_pack_theta(const float* w, unsigned short* o) {
  int idx = blockIdx.x * 256 + threadIdx.x; if (idx >= 192 * 64) return;
  int r = idx >> 6, i = idx & 63, k = r >> 6, oo = r & 63;
  o[idx] = f2bf(w[i * 192 + oo * 3 + k]);
}
// w2 (64,64,3) -> w2r[c2][kt*64+i]
__global__ void k_pack_w2(const float* w, unsigned short* o) {
  int idx = blockIdx.x * 256 + threadIdx.x; if (idx >= 64 * 192) return;
  int c2 = idx / 192, k = idx % 192, kt = k >> 6, i = k & 63;
  o[idx] = f2bf(w[c2 * 192 + i * 3 + kt]);
}
__global__ __launch_bounds__(256) void k_dsum(const float* __restrict__ A, float* __restrict__ d) {
  __shared__ float red[256];
  int row = blockIdx.x; float s = 0.f;
  for (int j = threadIdx.x; j < NN; j += 256) s += A[(long)row * NN + j];
  red[threadIdx.x] = s; __syncthreads();
  for (int off = 128; off; off >>= 1) { if (threadIdx.x < off) red[threadIdx.x] += red[threadIdx.x + off]; __syncthreads(); }
  if (!threadIdx.x) d[row] = red[0];
}
__global__ void k_buildL0(const float* __restrict__ A, const float* __restrict__ d, float* __restrict__ L0) {
  int idx = blockIdx.x * 256 + threadIdx.x; if (idx >= NN * NN) return;
  int i = idx >> 10, j = idx & 1023;
  float di = d[i], dj = d[j];
  float v = ((i == j) ? di : 0.f) - A[idx];
  if (di > 0.f && dj > 0.f) v /= sqrtf(di * dj);
  L0[idx] = v;
}
// power iteration for lambda_max of symmetric L0 (single workgroup)
__global__ __launch_bounds__(1024) void k_power(const float* __restrict__ L0, float* __restrict__ lamOut) {
  __shared__ float vsh[NN];
  __shared__ float red[32];
  const int i = threadIdx.x;
  vsh[i] = 0.03125f;   // 1/sqrt(1024)
  __syncthreads();
  float nrm2 = 1.f;
  for (int it = 0; it < 96; ++it) {
    float acc = 0.f;
    for (int j = 0; j < NN; ++j) acc += L0[(long)j * NN + i] * vsh[j];  // symmetric -> coalesced
    float s = acc * acc;
    for (int off = 16; off; off >>= 1) s += __shfl_down(s, off, 32);
    if ((i & 31) == 0) red[i >> 5] = s;
    __syncthreads();
    if (i < 32) {
      float t = red[i];
      for (int off = 16; off; off >>= 1) t += __shfl_down(t, off, 32);
      if (i == 0) red[0] = t;
    }
    __syncthreads();
    nrm2 = red[0];
    const float inv = rsqrtf(fmaxf(nrm2, 1e-30f));
    __syncthreads();
    vsh[i] = acc * inv;
    __syncthreads();
  }
  if (i == 0) lamOut[0] = sqrtf(fmaxf(nrm2, 1e-30f));
}
// Ls = 2*L0/lam - I  ->  bf16 into Lcat rows [0,1024)
__global__ void k_build_ls(const float* __restrict__ L0, const float* __restrict__ lam,
                           unsigned short* __restrict__ Lcat) {
  int idx = blockIdx.x * 256 + threadIdx.x; if (idx >= NN * NN) return;
  int i = idx >> 10, j = idx & 1023;
  Lcat[idx] = f2bf(2.f * L0[idx] / lam[0] - ((i == j) ? 1.f : 0.f));
}
// GLU: xt1[g][i][m] = (xc[g][i][m] + x[b,i,t+2,m]) * sigmoid(xc[g][64+i][m])
__global__ void k_glu(const float* __restrict__ xc, const float* __restrict__ x,
                      unsigned short* __restrict__ xt1) {
  int idx = blockIdx.x * 256 + threadIdx.x; if (idx >= G1C * CC * NN) return;
  int m = idx & 1023, i = (idx >> 10) & 63, g = idx >> 16;
  int b = g / T1, t = g % T1;
  float P = xc[(long)g * 131072 + (long)i * NN + m];
  float Q = xc[(long)g * 131072 + (long)(i + 64) * NN + m];
  float r = x[((long)(b * CC + i) * TT + (t + 2)) * NN + m];
  xt1[idx] = f2bf((P + r) * (1.f / (1.f + __expf(-Q))));
}
__global__ __launch_bounds__(256) void k_ln_stats(const float* __restrict__ xt2,
                                                  float* __restrict__ mu, float* __restrict__ rstd) {
  __shared__ float rs[256], rq[256];
  const int g = blockIdx.x;
  float s = 0.f, q = 0.f;
  const float* p = xt2 + (long)g * 65536;
  for (int j = threadIdx.x; j < 65536; j += 256) { float v = p[j]; s += v; q += v * v; }
  rs[threadIdx.x] = s; rq[threadIdx.x] = q; __syncthreads();
  for (int off = 128; off; off >>= 1) {
    if (threadIdx.x < off) { rs[threadIdx.x] += rs[threadIdx.x + off]; rq[threadIdx.x] += rq[threadIdx.x + off]; }
    __syncthreads();
  }
  if (!threadIdx.x) {
    float m = rs[0] * (1.f / 65536.f);
    float v = rq[0] * (1.f / 65536.f) - m * m;
    mu[g] = m; rstd[g] = rsqrtf(v + 1e-5f);
  }
}
__global__ void k_ln_apply(const float* __restrict__ xt2, const float* __restrict__ mu,
                           const float* __restrict__ rstd, const float* __restrict__ gamma,
                           const float* __restrict__ beta, float* __restrict__ out) {
  int idx = blockIdx.x * 256 + threadIdx.x; if (idx >= NBAT * CC * T2 * NN) return;
  int n = idx & 1023, rem = idx >> 10;
  int t = rem % T2; rem /= T2;
  int c = rem & 63, b = rem >> 6;
  int g = b * T2 + t;
  float v = xt2[(long)g * 65536 + (long)c * NN + n];
  out[idx] = (v - mu[g]) * rstd[g] * gamma[n * 64 + c] + beta[n * 64 + c];
}

// ---------------- orchestration ----------------
extern "C" void kernel_launch(void* const* d_in, const int* in_sizes, int n_in,
                              void* d_out, int out_size, void* d_ws, size_t ws_size,
                              hipStream_t stream) {
  const float* x      = (const float*)d_in[0];
  const float* A      = (const float*)d_in[1];
  const float* w1     = (const float*)d_in[2];
  const float* b1     = (const float*)d_in[3];
  const float* theta  = (const float*)d_in[4];
  const float* btheta = (const float*)d_in[5];
  const float* w2     = (const float*)d_in[6];
  const float* b2     = (const float*)d_in[7];
  const float* gamma  = (const float*)d_in[8];
  const float* beta   = (const float*)d_in[9];
  float* out = (float*)d_out;
  (void)in_sizes; (void)n_in; (void)out_size; (void)ws_size;

  char* w = (char*)d_ws;
  size_t off = 0;
  auto alloc = [&](size_t bytes) -> char* {
    char* p = w + off; off = (off + bytes + 255) & ~(size_t)255; return p;
  };
  float* dvec            = (float*)alloc((size_t)NN * 4);
  float* lam             = (float*)alloc(256);
  float* L0              = (float*)alloc((size_t)NN * NN * 4);
  unsigned short* Lcat   = (unsigned short*)alloc((size_t)2048 * NN * 2);     // [Ls;L2] row-major
  unsigned short* w1r    = (unsigned short*)alloc((size_t)128 * 192 * 2);
  unsigned short* thetaE = (unsigned short*)alloc((size_t)192 * 64 * 2);
  unsigned short* w2r    = (unsigned short*)alloc((size_t)64 * 192 * 2);
  unsigned short* x_bf   = (unsigned short*)alloc((size_t)NBAT * CC * TT * NN * 2);
  float* xc              = (float*)alloc((size_t)G1C * 128 * NN * 4);
  unsigned short* xt1    = (unsigned short*)alloc((size_t)G1C * CC * NN * 2); // [g][i][m]
  unsigned short* Ybuf   = (unsigned short*)alloc((size_t)G1C * CC * 3072 * 2); // [g][o][k*1024+m]
  unsigned short* xs     = (unsigned short*)alloc((size_t)G1C * CC * NN * 2); // [g][i][n]
  float* xt2             = (float*)alloc((size_t)G2C * CC * NN * 4);          // [g][c2][n]
  float* mu              = (float*)alloc((size_t)G2C * 4);
  float* rstd            = (float*)alloc((size_t)G2C * 4);

  // 1. pack weights + x->bf16
  hipLaunchKernelGGL(k_pack_w1,    dim3(96),  dim3(256), 0, stream, w1, w1r);
  hipLaunchKernelGGL(k_pack_theta, dim3(48),  dim3(256), 0, stream, theta, thetaE);
  hipLaunchKernelGGL(k_pack_w2,    dim3(48),  dim3(256), 0, stream, w2, w2r);
  {
    int n = NBAT * CC * TT * NN;
    hipLaunchKernelGGL(k_f32_to_bf16, dim3((n + 255) / 256), dim3(256), 0, stream, x, x_bf, n);
  }
  // 2. Laplacian: d, L0, lambda_max, Ls
  hipLaunchKernelGGL(k_dsum,    dim3(NN),   dim3(256), 0, stream, A, dvec);
  hipLaunchKernelGGL(k_buildL0, dim3(4096), dim3(256), 0, stream, A, dvec, L0);
  hipLaunchKernelGGL(k_power,   dim3(1),    dim3(1024), 0, stream, L0, lam);
  hipLaunchKernelGGL(k_build_ls,dim3(4096), dim3(256), 0, stream, L0, lam, Lcat);

  // 3. L2 = 2*Ls@Ls - I  (WMMA; B uses symmetry of Ls) -> Lcat rows [1024,2048)
  {
    GemmParams p{};
    p.A = Lcat; p.aGT = 1; p.aRS = 1024;
    p.B = Lcat; p.bGT = 1; p.bShift = 10; p.bChunk = 0; p.bKS = 1024;
    p.Cb = Lcat; p.cGT = 1; p.cCn = (long)1024 * 1024; p.cD = LINF; p.cS2 = 1024; p.cSC = 1;
    p.M = 1024; p.K = 1024; p.scaleEye = 1;
    hipLaunchKernelGGL(gemm_bf16_wmma, dim3(8, 16, 1), dim3(256), 0, stream, p);
  }
  // 4. conv1: xc[g][co][m] = w1r @ patch(x_bf) + b1
  {
    GemmParams p{};
    p.A = w1r; p.aGT = 1; p.aRS = 192;
    p.B = x_bf; p.bGT = T1; p.bBS = (long)CC * TT * NN; p.bTS = NN;
    p.bShift = 6; p.bChunk = NN; p.bKS = (long)TT * NN;   // k=(kt*64+i): kt*N + i*T*N
    p.Cf = xc; p.cGT = 1; p.cTS = (long)128 * NN; p.cD = LINF; p.cS2 = NN; p.cSC = 1;
    p.bias = b1; p.M = 128; p.K = 192;
    hipLaunchKernelGGL(gemm_bf16_wmma, dim3(8, 2, G1C), dim3(256), 0, stream, p);
  }
  // 5. GLU -> xt1 bf16
  {
    int n = G1C * CC * NN;
    hipLaunchKernelGGL(k_glu, dim3((n + 255) / 256), dim3(256), 0, stream, xc, x, xt1);
  }
  // 6. channel mix: Y[g][o][k*1024+m] = theta^T xt1
  {
    GemmParams p{};
    p.A = thetaE; p.aGT = 1; p.aRS = 64;
    p.B = xt1; p.bGT = 1; p.bTS = (long)CC * NN;
    p.bShift = 6; p.bChunk = 0; p.bKS = NN;               // k=i: i*1024
    p.Cb = Ybuf; p.cGT = 1; p.cTS = (long)CC * 3072;
    p.cD = 64; p.cS1 = 1024; p.cS2 = 3072; p.cSC = 1;     // row (k*64+o) -> o*3072 + k*1024
    p.M = 192; p.K = 64;
    hipLaunchKernelGGL(gemm_bf16_wmma, dim3(8, 3, G1C), dim3(256), 0, stream, p);
  }
  // 7. graph mix (dominant GEMM): xs = relu(Y[:,1024:] @ [Ls;L2] + b_theta + Y0 + xt1)
  {
    GemmParams p{};
    p.A = Ybuf; p.aGT = 1; p.aTS = (long)CC * 3072; p.aCn = 1024; p.aRS = 3072;
    p.B = Lcat; p.bGT = 1; p.bShift = 11; p.bChunk = 0; p.bKS = 1024;
    p.Cb = xs; p.cGT = 1; p.cTS = (long)CC * NN; p.cD = LINF; p.cS2 = NN; p.cSC = 1;
    p.bias = btheta;
    p.R1 = xt1;  p.r1GT = 1; p.r1TS = (long)CC * NN;  p.r1RS = NN;   p.r1CS = 1;
    p.R2 = Ybuf; p.r2GT = 1; p.r2TS = (long)CC * 3072; p.r2RS = 3072; p.r2CS = 1; // Y0 identity term
    p.M = 64; p.K = 2048; p.act = 1;
    hipLaunchKernelGGL(gemm_bf16_wmma, dim3(8, 1, G1C), dim3(256), 0, stream, p);
  }
  // 8. conv2: xt2 = relu(w2r @ patch(xs) + b2 + xs[:,t+2])
  {
    GemmParams p{};
    p.A = w2r; p.aGT = 1; p.aRS = 192;
    p.B = xs; p.bGT = T2; p.bBS = (long)T1 * CC * NN; p.bTS = (long)CC * NN;
    p.bShift = 6; p.bChunk = (long)CC * NN; p.bKS = NN;   // kt*65536 + i*1024
    p.Cf = xt2; p.cGT = 1; p.cTS = (long)CC * NN; p.cD = LINF; p.cS2 = NN; p.cSC = 1;
    p.bias = b2;
    p.R1 = xs; p.r1GT = T2; p.r1BS = (long)T1 * CC * NN; p.r1TS = (long)CC * NN;
    p.r1Cn = (long)2 * CC * NN; p.r1RS = NN; p.r1CS = 1;
    p.M = 64; p.K = 192; p.act = 1;
    hipLaunchKernelGGL(gemm_bf16_wmma, dim3(8, 1, G2C), dim3(256), 0, stream, p);
  }
  // 9. LayerNorm over (N,C) per (b,t) + final transpose to (B,C,T,N)
  hipLaunchKernelGGL(k_ln_stats, dim3(G2C), dim3(256), 0, stream, xt2, mu, rstd);
  {
    int n = NBAT * CC * T2 * NN;
    hipLaunchKernelGGL(k_ln_apply, dim3((n + 255) / 256), dim3(256), 0, stream,
                       xt2, mu, rstd, gamma, beta, out);
  }
}
